// DiffusionGPT_65335042507219
// MI455X (gfx1250) — matrix operand order repaired and checked by reading, not compiled, not run on hardware
//
#include <hip/hip_runtime.h>
#include <hip/hip_bf16.h>
#include <math.h>

typedef __bf16 bf16;
typedef __attribute__((ext_vector_type(16))) __bf16        v16bf;
typedef __attribute__((ext_vector_type(8)))  float         v8f;
typedef __attribute__((ext_vector_type(8)))  unsigned int  v8u;

#define WMMA_BF16(a, b, c) \
  __builtin_amdgcn_wmma_f32_16x16x32_bf16(false, (a), false, (b), (short)0, (c), false, false)

// ---- cross-lane helpers: DPP16 row_xmask butterflies (pure VALU, no LDS) ----
template <int CTRL>
__device__ __forceinline__ float dpp_movf(float x) {
  return __builtin_bit_cast(
      float, __builtin_amdgcn_update_dpp(0, __builtin_bit_cast(int, x), CTRL, 0xF, 0xF, true));
}
// max/add across each 16-lane group (row_xmask 1,2,4,8)
__device__ __forceinline__ float redmax16(float x) {
  x = fmaxf(x, dpp_movf<0x161>(x));
  x = fmaxf(x, dpp_movf<0x162>(x));
  x = fmaxf(x, dpp_movf<0x164>(x));
  x = fmaxf(x, dpp_movf<0x168>(x));
  return x;
}
__device__ __forceinline__ float redadd16(float x) {
  x += dpp_movf<0x161>(x);
  x += dpp_movf<0x162>(x);
  x += dpp_movf<0x164>(x);
  x += dpp_movf<0x168>(x);
  return x;
}
// swap the two 16-lane rows of a wave32 (identity selects)
__device__ __forceinline__ float xswap16(float x) {
  int xi = __builtin_bit_cast(int, x);
  int r = __builtin_amdgcn_permlanex16(xi, xi, 0x76543210, 0xfedcba98, false, false);
  return __builtin_bit_cast(float, r);
}
__device__ __forceinline__ float redadd32(float x) {
  x = redadd16(x);
  return x + xswap16(x);
}

// ---- fragment helpers (contiguous k = 16*half + s, same for A and B) ----------------------
__device__ __forceinline__ unsigned int ld_u32_bf(const bf16* p) {
  unsigned int r;
  __builtin_memcpy(&r, p, 4);
  return r;
}

// A-matrix 16x32 fragment, per-lane: caller passes p = rowBase + 16*halfLane.
__device__ __forceinline__ v16bf fragA16(const bf16* p) {
  v8u u;
#pragma unroll
  for (int v = 0; v < 4; ++v) u[v] = ld_u32_bf(p + 2 * v);
#pragma unroll
  for (int v = 0; v < 4; ++v) u[4 + v] = ld_u32_bf(p + 8 + 2 * v);
  return __builtin_bit_cast(v16bf, u);
}

// B-matrix 32x16 fragment, per-lane data is 16 contiguous bf16
// (caller passes p = base + 16*halfLane).
__device__ __forceinline__ v16bf fragB16(const bf16* p) {
  v8u u;
#pragma unroll
  for (int v = 0; v < 8; ++v) u[v] = ld_u32_bf(p + 2 * v);
  return __builtin_bit_cast(v16bf, u);
}

// ---- elementwise converts ---------------------------------------------------
__global__ void f2bf_kernel(const float* __restrict__ in, bf16* __restrict__ out, int n) {
  int i = blockIdx.x * 256 + threadIdx.x;
  if (i < n) out[i] = (bf16)in[i];
}

// ---- token / goal / sigma embedding ----------------------------------------
__global__ void embed_kernel(const float* __restrict__ sa, const float* __restrict__ goals,
                             const float* __restrict__ sigma, const float* __restrict__ sigma_w,
                             const float* __restrict__ sigma_b, const float* __restrict__ tok_w,
                             const float* __restrict__ tok_b, const float* __restrict__ goal_w,
                             const float* __restrict__ goal_b, const float* __restrict__ pos,
                             float* __restrict__ x, int Bn, int S, int E, int T, int SA, int G) {
  size_t idx = (size_t)blockIdx.x * 256 + threadIdx.x;
  size_t tot = (size_t)Bn * S * E;
  if (idx >= tot) return;
  int e = (int)(idx % E);
  int s = (int)((idx / E) % S);
  int b = (int)(idx / ((size_t)E * S));
  float v;
  if (s == 0) {
    v = (logf(sigma[b]) * 0.25f) * sigma_w[e] + sigma_b[e];
  } else if (s == 1) {
    float acc = goal_b[e];
    for (int g = 0; g < G; ++g) acc += goals[b * G + g] * goal_w[g * E + e];
    v = acc + pos[e];  // pos row 0
  } else {
    float acc = tok_b[e];
    const float* sap = sa + ((size_t)b * T + (s - 2)) * SA;
#pragma unroll 8
    for (int i = 0; i < SA; ++i) acc += sap[i] * tok_w[i * E + e];
    v = acc + pos[(size_t)(s - 1) * E + e];
  }
  x[idx] = v;
}

// ---- LayerNorm (one wave per row of E=512), f32 in -> bf16 out -------------
__global__ __launch_bounds__(256) void ln_kernel(const float* __restrict__ x,
                                                 const float* __restrict__ g,
                                                 const float* __restrict__ bt,
                                                 bf16* __restrict__ out, int E) {
  const int lane = threadIdx.x & 31, wid = threadIdx.x >> 5;
  const size_t row = (size_t)blockIdx.x * 8 + wid;
  const float* xr = x + row * E;
  float v[16], s = 0.f;
#pragma unroll
  for (int i = 0; i < 16; ++i) { v[i] = xr[lane + i * 32]; s += v[i]; }
  s = redadd32(s);
  const float mean = s * (1.f / (float)E);
  float q = 0.f;
#pragma unroll
  for (int i = 0; i < 16; ++i) { float t = v[i] - mean; q += t * t; }
  q = redadd32(q);
  const float rstd = rsqrtf(q * (1.f / (float)E) + 1e-5f);
  bf16* orow = out + row * E;
#pragma unroll
  for (int i = 0; i < 16; ++i) {
    int c = lane + i * 32;
    orow[c] = (bf16)((v[i] - mean) * rstd * g[c] + bt[c]);
  }
}

// ---- WMMA GEMM: C = A(MxK,bf16) * B(KxN,bf16) + bias (+resid)(+gelu) -------
// Block tile 128x128x32, 256 threads = 8 waves in a 4x2 grid, each wave 32x64.
template <bool OUTBF, bool RESID, bool GELU_ACT, bool SHIFT2>
__global__ __launch_bounds__(256) void gemm_wmma(
    const bf16* __restrict__ A, const bf16* __restrict__ Bm,
    const float* __restrict__ bias, const float* __restrict__ resid,
    float* __restrict__ outF, bf16* __restrict__ outB,
    int M, int N, int K, int S_tok, int T_tok) {
  __shared__ bf16 As[128][36];            // row-major A tile, padded
  __shared__ unsigned int Bs[16][132];    // pair-packed B tile: Bs[k/2][n] = {k,k+1}
  const int tid = threadIdx.x;
  const int lane = tid & 31, wid = tid >> 5;
  const int half = lane >> 4, l16 = lane & 15;
  const int wm = wid >> 1, wn = wid & 1;
  const int m0 = blockIdx.y * 128, n0 = blockIdx.x * 128;

  v8f acc[2][4];
#pragma unroll
  for (int a = 0; a < 2; ++a)
#pragma unroll
    for (int b = 0; b < 4; ++b)
#pragma unroll
      for (int j = 0; j < 8; ++j) acc[a][b][j] = 0.f;

  const int arow = tid >> 1, akh = tid & 1;
  for (int k0 = 0; k0 < K; k0 += 32) {
    {  // A tile: 2 threads per row, 16 bf16 each (two b128 loads)
      const bf16* gp = A + (size_t)(m0 + arow) * K + k0 + akh * 16;
      uint4 p0, p1;
      __builtin_memcpy(&p0, gp, 16);
      __builtin_memcpy(&p1, gp + 8, 16);
      unsigned int* dst = (unsigned int*)&As[arow][akh * 16];
      dst[0] = p0.x; dst[1] = p0.y; dst[2] = p0.z; dst[3] = p0.w;
      dst[4] = p1.x; dst[5] = p1.y; dst[6] = p1.z; dst[7] = p1.w;
    }
#pragma unroll
    for (int j = 0; j < 8; ++j) {  // B tile, pair-packed for 1-op fragment loads
      int e = tid + j * 256;
      int n = e & 127, kp = e >> 7;
      int gn = n0 + n;
      unsigned int lo = 0, hi = 0;
      if (gn < N) {
        lo = (unsigned int)__builtin_bit_cast(unsigned short, Bm[(size_t)(k0 + 2 * kp) * N + gn]);
        hi = (unsigned int)__builtin_bit_cast(unsigned short, Bm[(size_t)(k0 + 2 * kp + 1) * N + gn]);
      }
      Bs[kp][n] = lo | (hi << 16);
    }
    if (k0 + 32 < K) {  // hint next tiles into cache (global_prefetch_b8)
      __builtin_prefetch(A + (size_t)(m0 + arow) * K + (k0 + 32) + akh * 16, 0, 1);
      __builtin_prefetch(Bm + (size_t)(k0 + 32) * N + n0 + (tid & 127), 0, 1);
    }
    __syncthreads();

    v16bf af[2], bfv[4];
#pragma unroll
    for (int mf = 0; mf < 2; ++mf)
      af[mf] = fragA16(&As[wm * 32 + mf * 16 + l16][0] + 16 * half);
#pragma unroll
    for (int nf = 0; nf < 4; ++nf) {
      v8u u;
      int col = wn * 64 + nf * 16 + l16;
#pragma unroll
      for (int v = 0; v < 8; ++v) u[v] = Bs[half * 8 + v][col];
      bfv[nf] = __builtin_bit_cast(v16bf, u);
    }
#pragma unroll
    for (int mf = 0; mf < 2; ++mf)
#pragma unroll
      for (int nf = 0; nf < 4; ++nf)
        acc[mf][nf] = WMMA_BF16(af[mf], bfv[nf], acc[mf][nf]);
    __syncthreads();
  }

  // epilogue — C layout: elem(m,n): vgpr j=m%8, lane=(m/8)*16 + n
#pragma unroll
  for (int mf = 0; mf < 2; ++mf) {
#pragma unroll
    for (int nf = 0; nf < 4; ++nf) {
      int gcol = n0 + wn * 64 + nf * 16 + l16;
      float bv = (gcol < N) ? bias[gcol] : 0.f;
#pragma unroll
      for (int j = 0; j < 8; ++j) {
        int grow = m0 + wm * 32 + mf * 16 + half * 8 + j;
        float v = acc[mf][nf][j] + bv;
        if (GELU_ACT) v = 0.5f * v * (1.f + erff(v * 0.70710678118f));
        if (gcol < N) {
          size_t oidx = (size_t)grow * N + gcol;
          if (RESID) v += resid[oidx];
          if (SHIFT2) {
            int bb = grow / S_tok, ss = grow % S_tok;
            if (ss >= 2) outF[((size_t)bb * T_tok + (ss - 2)) * N + gcol] = v;
          } else if (OUTBF) {
            outB[oidx] = (bf16)v;
          } else {
            outF[oidx] = v;
          }
        }
      }
    }
  }
}

// ---- causal flash attention: one wave per 16-query tile, D=64 --------------
__global__ __launch_bounds__(256) void attn_kernel(const bf16* __restrict__ Q,
                                                   const bf16* __restrict__ Kt,
                                                   const bf16* __restrict__ Vt,
                                                   bf16* __restrict__ Y,
                                                   int S, int E, int H, float scale) {
  __shared__ bf16 Pl[8][16 * 36];  // per-wave P transpose staging (C-layout -> A-layout)
  const int lane = threadIdx.x & 31, wid = threadIdx.x >> 5;
  const int half = lane >> 4, l16 = lane & 15;
  const int nqc = S >> 7;                       // 128 query rows per workgroup
  const int bh = blockIdx.x / nqc, qc = blockIdx.x % nqc;
  const int b = bh / H, h = bh % H;
  const int qr = (qc * 8 + wid) * 16;           // this wave's query-tile base row
  const int Dh = 64;

  // Q resident as two A fragments (d chunks 0..31, 32..63)
  v16bf aQ[2];
  {
    const bf16* qp = Q + ((size_t)(b * S + qr + l16)) * E + h * Dh + 16 * half;
    aQ[0] = fragA16(qp);
    aQ[1] = fragA16(qp + 32);
  }

  v8f o[4];
#pragma unroll
  for (int t = 0; t < 4; ++t)
#pragma unroll
    for (int j = 0; j < 8; ++j) o[t][j] = 0.f;
  float rmax[8], rsum[8];
#pragma unroll
  for (int j = 0; j < 8; ++j) { rmax[j] = -1e30f; rsum[j] = 0.f; }

  bf16* Pw = &Pl[wid][0];
  const int nkb = (qr + 47) >> 5;  // causal: 32-key blocks covering keys 0..qr+15
  for (int kb = 0; kb < nkb; ++kb) {
    // scores: two 16x16 C-tiles (key halves), K-dim = D in two 32-chunks
    v8f c0, c1;
#pragma unroll
    for (int j = 0; j < 8; ++j) { c0[j] = 0.f; c1[j] = 0.f; }
    {
      const bf16* kp0 = Kt + ((size_t)(b * S + kb * 32 + l16)) * E + h * Dh + 16 * half;
      c0 = WMMA_BF16(aQ[0], fragB16(kp0), c0);
      c0 = WMMA_BF16(aQ[1], fragB16(kp0 + 32), c0);
      const bf16* kp1 = kp0 + (size_t)16 * E;
      c1 = WMMA_BF16(aQ[0], fragB16(kp1), c1);
      c1 = WMMA_BF16(aQ[1], fragB16(kp1 + 32), c1);
    }
    // online softmax in C layout: row stats per (vgpr j, lane-half), replicated
    // over each 16-lane group; reductions are pure-VALU DPP row_xmask butterflies.
#pragma unroll
    for (int j = 0; j < 8; ++j) {
      const int row = qr + half * 8 + j;
      const int col0 = kb * 32 + l16, col1 = col0 + 16;
      float s0 = c0[j] * scale, s1 = c1[j] * scale;
      if (col0 > row) s0 = -1e30f;
      if (col1 > row) s1 = -1e30f;
      float mx = redmax16(fmaxf(s0, s1));
      float mn = fmaxf(rmax[j], mx);
      float f = __expf(rmax[j] - mn);
      rmax[j] = mn;
      float p0 = __expf(s0 - mn), p1 = __expf(s1 - mn);
      float ps = redadd16(p0 + p1);
      rsum[j] = rsum[j] * f + ps;
#pragma unroll
      for (int t = 0; t < 4; ++t) o[t][j] *= f;
      Pw[(half * 8 + j) * 36 + l16] = (bf16)p0;
      Pw[(half * 8 + j) * 36 + 16 + l16] = (bf16)p1;
    }
    // per-wave LDS RAW fence (trip counts differ per wave -> no __syncthreads)
    asm volatile("s_wait_dscnt 0" ::: "memory");
    v16bf aP = fragA16(Pw + l16 * 36 + 16 * half);
#pragma unroll
    for (int t = 0; t < 4; ++t) {  // PV: B frag of V (32 keys x 16 dout)
      const bf16* vp = Vt + ((size_t)(b * S + kb * 32 + half * 16)) * E + h * Dh + t * 16 + l16;
      v8u u;
#pragma unroll
      for (int v = 0; v < 8; ++v) {
        unsigned int lo = (unsigned int)__builtin_bit_cast(unsigned short, vp[(size_t)(2 * v) * E]);
        unsigned int hi = (unsigned int)__builtin_bit_cast(unsigned short, vp[(size_t)(2 * v + 1) * E]);
        u[v] = lo | (hi << 16);
      }
      o[t] = WMMA_BF16(aP, __builtin_bit_cast(v16bf, u), o[t]);
    }
  }
#pragma unroll
  for (int j = 0; j < 8; ++j) {
    const float inv = __builtin_amdgcn_rcpf(rsum[j]);
    const int row = qr + half * 8 + j;
#pragma unroll
    for (int t = 0; t < 4; ++t)
      Y[((size_t)(b * S + row)) * E + h * Dh + t * 16 + l16] = (bf16)(o[t][j] * inv);
  }
}

// ---------------------------------------------------------------------------
extern "C" void kernel_launch(void* const* d_in, const int* in_sizes, int n_in,
                              void* d_out, int out_size, void* d_ws, size_t ws_size,
                              hipStream_t stream) {
  (void)in_sizes; (void)n_in; (void)out_size; (void)ws_size;
  const int Bn = 8, T = 1022, S = 1024, E = 512, F = 2048, L = 4, H = 8, SA = 72, G = 3;
  const float scale = 0.125f;  // 1/sqrt(64)

  const float* sa      = (const float*)d_in[0];
  const float* goals   = (const float*)d_in[1];
  const float* sigma   = (const float*)d_in[2];
  const float* sigma_w = (const float*)d_in[3];
  const float* sigma_b = (const float*)d_in[4];
  const float* tok_w   = (const float*)d_in[5];
  const float* tok_b   = (const float*)d_in[6];
  const float* goal_w  = (const float*)d_in[7];
  const float* goal_b  = (const float*)d_in[8];
  const float* pos     = (const float*)d_in[9];
  const float* ln1_g   = (const float*)d_in[10];
  const float* ln1_b   = (const float*)d_in[11];
  const float* q_w     = (const float*)d_in[12];
  const float* q_b     = (const float*)d_in[13];
  const float* k_w     = (const float*)d_in[14];
  const float* k_b     = (const float*)d_in[15];
  const float* v_w     = (const float*)d_in[16];
  const float* v_b     = (const float*)d_in[17];
  const float* proj_w  = (const float*)d_in[18];
  const float* proj_b  = (const float*)d_in[19];
  const float* ln2_g   = (const float*)d_in[20];
  const float* ln2_b   = (const float*)d_in[21];
  const float* mlp_w1  = (const float*)d_in[22];
  const float* mlp_b1  = (const float*)d_in[23];
  const float* mlp_w2  = (const float*)d_in[24];
  const float* mlp_b2  = (const float*)d_in[25];
  const float* lnf_g   = (const float*)d_in[26];
  const float* lnf_b   = (const float*)d_in[27];
  const float* pred_w  = (const float*)d_in[28];
  const float* pred_b  = (const float*)d_in[29];

  char* wsp = (char*)d_ws;
  size_t off = 0;
  auto alloc = [&](size_t bytes) -> void* {
    void* p = wsp + off;
    off += (bytes + 255) & ~(size_t)255;
    return p;
  };
  const size_t MR = (size_t)Bn * S;  // 8192 rows
  float* x    = (float*)alloc(MR * E * sizeof(float));
  bf16* hbuf  = (bf16*)alloc(MR * E * 2);
  bf16* qbuf  = (bf16*)alloc(MR * E * 2);
  bf16* kbuf  = (bf16*)alloc(MR * E * 2);
  bf16* vbuf  = (bf16*)alloc(MR * E * 2);
  bf16* ybuf  = (bf16*)alloc(MR * E * 2);
  bf16* mid   = (bf16*)alloc(MR * F * 2);
  bf16* wq    = (bf16*)alloc((size_t)L * E * E * 2);
  bf16* wk    = (bf16*)alloc((size_t)L * E * E * 2);
  bf16* wv    = (bf16*)alloc((size_t)L * E * E * 2);
  bf16* wp    = (bf16*)alloc((size_t)L * E * E * 2);
  bf16* w1    = (bf16*)alloc((size_t)L * E * F * 2);
  bf16* w2    = (bf16*)alloc((size_t)L * F * E * 2);
  bf16* wpr   = (bf16*)alloc((size_t)E * SA * 2);

  auto conv = [&](const float* src, bf16* dst, int n) {
    f2bf_kernel<<<(n + 255) / 256, 256, 0, stream>>>(src, dst, n);
  };
  conv(q_w, wq, L * E * E);
  conv(k_w, wk, L * E * E);
  conv(v_w, wv, L * E * E);
  conv(proj_w, wp, L * E * E);
  conv(mlp_w1, w1, L * E * F);
  conv(mlp_w2, w2, L * F * E);
  conv(pred_w, wpr, E * SA);

  {
    size_t tot = MR * E;
    embed_kernel<<<(unsigned)((tot + 255) / 256), 256, 0, stream>>>(
        sa, goals, sigma, sigma_w, sigma_b, tok_w, tok_b, goal_w, goal_b, pos,
        x, Bn, S, E, T, SA, G);
  }

  dim3 blk(256);
  const int Mtiles = (int)(MR / 128);  // 64
  for (int l = 0; l < L; ++l) {
    ln_kernel<<<(unsigned)(MR / 8), 256, 0, stream>>>(x, ln1_g + l * E, ln1_b + l * E, hbuf, E);
    gemm_wmma<true, false, false, false><<<dim3(E / 128, Mtiles), blk, 0, stream>>>(
        hbuf, wq + (size_t)l * E * E, q_b + l * E, nullptr, nullptr, qbuf, (int)MR, E, E, S, T);
    gemm_wmma<true, false, false, false><<<dim3(E / 128, Mtiles), blk, 0, stream>>>(
        hbuf, wk + (size_t)l * E * E, k_b + l * E, nullptr, nullptr, kbuf, (int)MR, E, E, S, T);
    gemm_wmma<true, false, false, false><<<dim3(E / 128, Mtiles), blk, 0, stream>>>(
        hbuf, wv + (size_t)l * E * E, v_b + l * E, nullptr, nullptr, vbuf, (int)MR, E, E, S, T);
    attn_kernel<<<Bn * H * (S / 128), 256, 0, stream>>>(qbuf, kbuf, vbuf, ybuf, S, E, H, scale);
    gemm_wmma<false, true, false, false><<<dim3(E / 128, Mtiles), blk, 0, stream>>>(
        ybuf, wp + (size_t)l * E * E, proj_b + l * E, x, x, nullptr, (int)MR, E, E, S, T);
    ln_kernel<<<(unsigned)(MR / 8), 256, 0, stream>>>(x, ln2_g + l * E, ln2_b + l * E, hbuf, E);
    gemm_wmma<true, false, true, false><<<dim3(F / 128, Mtiles), blk, 0, stream>>>(
        hbuf, w1 + (size_t)l * E * F, mlp_b1 + l * F, nullptr, nullptr, mid, (int)MR, F, E, S, T);
    gemm_wmma<false, true, false, false><<<dim3(E / 128, Mtiles), blk, 0, stream>>>(
        mid, w2 + (size_t)l * F * E, mlp_b2 + l * E, x, x, nullptr, (int)MR, E, F, S, T);
  }
  ln_kernel<<<(unsigned)(MR / 8), 256, 0, stream>>>(x, lnf_g, lnf_b, hbuf, E);
  gemm_wmma<false, false, false, true><<<dim3(1, Mtiles), blk, 0, stream>>>(
      hbuf, wpr, pred_b, nullptr, (float*)d_out, nullptr, (int)MR, SA, E, S, T);
}
